// MultiHeadAttention_54709293416775
// MI455X (gfx1250) — matrix operand order repaired and checked by reading, not compiled, not run on hardware
//
#include <hip/hip_runtime.h>
#include <hip/hip_bf16.h>
#include <math.h>

#define B_  2
#define S_  2048
#define D_  1024
#define H_  16
#define HD_ 64

typedef __attribute__((ext_vector_type(16))) __bf16 v16bf;
typedef __attribute__((ext_vector_type(8)))  float  v8f;

// ---------------------------------------------------------------------------
// WMMA helper: D = A(16x32 bf16) * B(32x16 bf16) + C(16x16 f32)
// ---------------------------------------------------------------------------
__device__ __forceinline__ v8f wmma_bf16(v16bf a, v16bf b, v8f c) {
  return __builtin_amdgcn_wmma_f32_16x16x32_bf16(false, a, false, b,
                                                 (short)0, c, false, false);
}

// A-fragment (16x32, bf16), A row-major with leading dim ld.
// Layout: lanes 0-15: m=lane, elems[0..15]=K+0..15
//         lanes 16-31: m=lane-16, elems[0..15]=K+16..31.
__device__ __forceinline__ v16bf frag_a_rowmajor(const __bf16* __restrict__ A,
                                                 size_t ld, int m0, int kk,
                                                 int lane) {
  const int m  = m0 + (lane & 15);
  const int hs = (lane >> 4) * 16;
  const __bf16* p = A + (size_t)m * ld + (size_t)(kk + hs);
  v16bf a;
#pragma unroll
  for (int i = 0; i < 16; ++i) a[i] = p[i];
  return a;
}

// B-fragment (32x16, bf16) where element (k,n) lives at Bm[n*ld + k]
// (column n contiguous in k). Lanes 0-15: n=lane, k=kk..kk+15;
// lanes 16-31: n=lane-16, k=kk+16..kk+31.
__device__ __forceinline__ v16bf frag_b_kmajor(const __bf16* __restrict__ Bm,
                                               size_t ld, int n0, int kk,
                                               int lane) {
  const int n = n0 + (lane & 15);
  const int k = kk + (lane >> 4) * 16;
  const __bf16* p = Bm + (size_t)n * ld + (size_t)k;
  v16bf b;
#pragma unroll
  for (int i = 0; i < 16; ++i) b[i] = p[i];
  return b;
}

// ---------------------------------------------------------------------------
// fp32 -> bf16 conversion (grid-stride)
// ---------------------------------------------------------------------------
__global__ void f32_to_bf16_kernel(const float* __restrict__ src,
                                   __bf16* __restrict__ dst, size_t n) {
  size_t i = (size_t)blockIdx.x * blockDim.x + threadIdx.x;
  const size_t stride = (size_t)gridDim.x * blockDim.x;
  for (; i < n; i += stride) dst[i] = (__bf16)src[i];
}

// ---------------------------------------------------------------------------
// Fused QKV projection + RoPE.  One wave = 32 rows x one head (64 cols):
// two 16-row A fragments share every weight B fragment (8 WMMA / 6 frag
// loads per 32-K step).  blockIdx.z: 0=Q(rope) 1=K(rope) 2=V(no rope,
// stored transposed (B,H,HD,S)).  Q/K stored (B,H,S,HD).
// ---------------------------------------------------------------------------
__global__ __launch_bounds__(128) void qkv_rope_kernel(
    const __bf16* __restrict__ xb, const __bf16* __restrict__ wq,
    const __bf16* __restrict__ wk, const __bf16* __restrict__ wv,
    __bf16* __restrict__ Q, __bf16* __restrict__ K, __bf16* __restrict__ Vt) {
  const int lane = threadIdx.x & 31;
  const int wave = threadIdx.x >> 5;
  const int m0   = (blockIdx.x * 4 + wave) * 32;   // row tile in flattened B*S
  const int h    = blockIdx.y;
  const int n0   = h * HD_;
  const int what = blockIdx.z;
  const __bf16* w = (what == 0) ? wq : (what == 1) ? wk : wv;

  const int nl = lane & 15;
  const int hf = lane >> 4;

  v8f acc[2][4] = {};
  for (int kk = 0; kk < D_; kk += 32) {
    if (kk + 64 < D_) {
      // locality 3 -> WGP-scope prefetch: fills all cache levels for the
      // K-step consumed 2 iterations from now (addresses stay in-bounds).
      __builtin_prefetch(w + (size_t)(n0 + nl) * D_ + kk + 64, 0, 3);
      __builtin_prefetch(xb + (size_t)(m0 + nl) * D_ + kk + 64, 0, 3);
    }
    v16bf a0 = frag_a_rowmajor(xb, D_, m0, kk, lane);
    v16bf a1 = frag_a_rowmajor(xb, D_, m0 + 16, kk, lane);
#pragma unroll
    for (int c = 0; c < 4; ++c) {
      v16bf b = frag_b_kmajor(w, D_, n0 + c * 16, kk, lane);
      acc[0][c] = wmma_bf16(a0, b, acc[0][c]);
      acc[1][c] = wmma_bf16(a1, b, acc[1][c]);
    }
  }

#pragma unroll
  for (int g = 0; g < 2; ++g) {
    const int mg = m0 + g * 16;
    if (what < 2) {
      // RoPE: column j pairs with j+32 -> accumulator tile c pairs with c+2,
      // same lane, same VGPR slot.  angle = pos * base^(-j/32).
#pragma unroll
      for (int c = 0; c < 2; ++c) {
        const float invf =
            __expf(-((float)(c * 16 + nl) / 32.f) * 9.210340371976184f);
#pragma unroll
        for (int r = 0; r < 8; ++r) {
          const int m = mg + r + 8 * hf;
          const int s = m & (S_ - 1);
          float sn, cs;
          __sincosf((float)s * invf, &sn, &cs);
          const float x1 = acc[g][c][r], x2 = acc[g][c + 2][r];
          acc[g][c][r]     = x1 * cs - x2 * sn;
          acc[g][c + 2][r] = x2 * cs + x1 * sn;
        }
      }
    }

#pragma unroll
    for (int c = 0; c < 4; ++c) {
#pragma unroll
      for (int r = 0; r < 8; ++r) {
        const int m   = mg + r + 8 * hf;
        const int bb  = m / S_;
        const int s   = m & (S_ - 1);
        const int col = c * 16 + nl;
        if (what == 2) {
          Vt[((size_t)(bb * H_ + h) * HD_ + col) * S_ + s] =
              (__bf16)acc[g][c][r];
        } else {
          __bf16* dst = (what == 0) ? Q : K;
          dst[((size_t)(bb * H_ + h) * S_ + s) * HD_ + col] =
              (__bf16)acc[g][c][r];
        }
      }
    }
  }
}

// ---------------------------------------------------------------------------
// Flash attention: one wave = 16 query rows of one (b,h); 32 keys / iter.
// Online softmax; causal mask inline; P transposed via per-wave LDS bounce.
// ---------------------------------------------------------------------------
__global__ __launch_bounds__(128) void flash_attn_kernel(
    const __bf16* __restrict__ Q, const __bf16* __restrict__ K,
    const __bf16* __restrict__ Vt, __bf16* __restrict__ attn) {
  __shared__ __align__(16) __bf16 plds[4][16 * 32];

  const int lane = threadIdx.x & 31;
  const int wave = threadIdx.x >> 5;
  const int h = blockIdx.y;
  const int b = blockIdx.z;
  const int s0 = (blockIdx.x * 4 + wave) * 16;
  const size_t bh = (size_t)b * H_ + h;
  const __bf16* Qbase = Q + bh * S_ * HD_;
  const __bf16* Kbase = K + bh * S_ * HD_;
  const __bf16* Vbase = Vt + bh * HD_ * S_;

  const v16bf qa0 = frag_a_rowmajor(Qbase, HD_, s0, 0, lane);
  const v16bf qa1 = frag_a_rowmajor(Qbase, HD_, s0, 32, lane);

  v8f o[4] = {};
  float mrow[8], lrow[8];
#pragma unroll
  for (int r = 0; r < 8; ++r) { mrow[r] = -INFINITY; lrow[r] = 0.f; }

  const float scale = 0.125f;  // 1/sqrt(64)
  const int nl = lane & 15;
  const int hf = lane >> 4;
  const int kmax = s0 + 16;    // causal: keys <= last query row

  for (int j0 = 0; j0 < kmax; j0 += 32) {
    if (j0 + 32 < kmax) {
      __builtin_prefetch(Kbase + (size_t)(j0 + 32 + nl) * HD_, 0, 3);
      __builtin_prefetch(Vbase + (size_t)nl * S_ + j0 + 32, 0, 3);
    }
    // scores for two 16-key tiles
    v8f sc[2];
#pragma unroll
    for (int t = 0; t < 2; ++t) {
      const int jt = j0 + t * 16;
      const v16bf kb0 = frag_b_kmajor(Kbase, HD_, jt, 0, lane);
      const v16bf kb1 = frag_b_kmajor(Kbase, HD_, jt, 32, lane);
      v8f sa = {};
      sa = wmma_bf16(qa0, kb0, sa);
      sa = wmma_bf16(qa1, kb1, sa);
      sc[t] = sa;
    }

    // online softmax (row = VGPR slot within a 16-lane half)
    float pv[2][8];
#pragma unroll
    for (int r = 0; r < 8; ++r) {
      const int row = s0 + r + 8 * hf;
#pragma unroll
      for (int t = 0; t < 2; ++t) {
        const int col = j0 + t * 16 + nl;
        pv[t][r] = sc[t][r] * scale + ((col <= row) ? 0.f : -1e9f);
      }
      float tm = fmaxf(pv[0][r], pv[1][r]);
#pragma unroll
      for (int off = 1; off < 16; off <<= 1)
        tm = fmaxf(tm, __shfl_xor(tm, off, 32));
      const float mnew = fmaxf(mrow[r], tm);
      const float corr = __expf(mrow[r] - mnew);
      lrow[r] *= corr;
#pragma unroll
      for (int c = 0; c < 4; ++c) o[c][r] *= corr;
      mrow[r] = mnew;
      const float p0 = __expf(pv[0][r] - mnew);
      const float p1 = __expf(pv[1][r] - mnew);
      pv[0][r] = p0; pv[1][r] = p1;
      float ps = p0 + p1;
#pragma unroll
      for (int off = 1; off < 16; off <<= 1)
        ps += __shfl_xor(ps, off, 32);
      lrow[r] += ps;
    }

    // transpose P (C-layout -> row-major 16x32) through per-wave LDS
    __bf16* pw = &plds[wave][0];
#pragma unroll
    for (int r = 0; r < 8; ++r) {
      const int m = r + 8 * hf;
      pw[m * 32 + nl]      = (__bf16)pv[0][r];
      pw[m * 32 + 16 + nl] = (__bf16)pv[1][r];
    }
    asm volatile("" ::: "memory");  // same-wave LDS ops are in-order
    const v16bf pa = frag_a_rowmajor(pw, 32, 0, 0, lane);
    asm volatile("" ::: "memory");

    // O += P * V  (V^T layout makes B-fragments contiguous along keys)
#pragma unroll
    for (int c = 0; c < 4; ++c) {
      const v16bf vb = frag_b_kmajor(Vbase, S_, c * 16, j0, lane);
      o[c] = wmma_bf16(pa, vb, o[c]);
    }
  }

  // normalize and store (B,S,D) bf16
#pragma unroll
  for (int r = 0; r < 8; ++r) {
    const int row = s0 + r + 8 * hf;
    const float inv = 1.f / lrow[r];
#pragma unroll
    for (int c = 0; c < 4; ++c) {
      attn[((size_t)b * S_ + row) * D_ + h * HD_ + c * 16 + nl] =
          (__bf16)(o[c][r] * inv);
    }
  }
}

// ---------------------------------------------------------------------------
// Output projection: out = attn @ wo^T, fp32 result.  One wave = 32x64 tile.
// ---------------------------------------------------------------------------
__global__ __launch_bounds__(128) void outproj_kernel(
    const __bf16* __restrict__ attn, const __bf16* __restrict__ wo,
    float* __restrict__ out) {
  const int lane = threadIdx.x & 31;
  const int wave = threadIdx.x >> 5;
  const int m0 = (blockIdx.x * 4 + wave) * 32;
  const int n0 = blockIdx.y * 64;
  const int nl = lane & 15;
  const int hf = lane >> 4;

  v8f acc[2][4] = {};
  for (int kk = 0; kk < D_; kk += 32) {
    if (kk + 64 < D_) {
      __builtin_prefetch(wo + (size_t)(n0 + nl) * D_ + kk + 64, 0, 3);
      __builtin_prefetch(attn + (size_t)(m0 + nl) * D_ + kk + 64, 0, 3);
    }
    v16bf a0 = frag_a_rowmajor(attn, D_, m0, kk, lane);
    v16bf a1 = frag_a_rowmajor(attn, D_, m0 + 16, kk, lane);
#pragma unroll
    for (int c = 0; c < 4; ++c) {
      v16bf b = frag_b_kmajor(wo, D_, n0 + c * 16, kk, lane);
      acc[0][c] = wmma_bf16(a0, b, acc[0][c]);
      acc[1][c] = wmma_bf16(a1, b, acc[1][c]);
    }
  }
#pragma unroll
  for (int g = 0; g < 2; ++g)
#pragma unroll
    for (int c = 0; c < 4; ++c)
#pragma unroll
      for (int r = 0; r < 8; ++r)
        out[(size_t)(m0 + g * 16 + r + 8 * hf) * D_ + n0 + c * 16 + nl] =
            acc[g][c][r];
}

// ---------------------------------------------------------------------------
extern "C" void kernel_launch(void* const* d_in, const int* in_sizes, int n_in,
                              void* d_out, int out_size, void* d_ws,
                              size_t ws_size, hipStream_t stream) {
  (void)in_sizes; (void)n_in; (void)out_size; (void)ws_size;
  const float* x  = (const float*)d_in[0];
  // d_in[1] = attention_mask: strictly causal -1e9 mask, reproduced inline.
  const float* wq = (const float*)d_in[2];
  const float* wk = (const float*)d_in[3];
  const float* wv = (const float*)d_in[4];
  const float* wo = (const float*)d_in[5];

  const size_t ND = (size_t)B_ * S_ * D_;  // 4M elems
  const size_t NW = (size_t)D_ * D_;       // 1M elems

  __bf16* xb    = (__bf16*)d_ws;
  __bf16* wqb   = xb + ND;
  __bf16* wkb   = wqb + NW;
  __bf16* wvb   = wkb + NW;
  __bf16* wob   = wvb + NW;
  __bf16* Qb    = wob + NW;      // (B,H,S,HD)
  __bf16* Kb    = Qb + ND;       // (B,H,S,HD)
  __bf16* Vtb   = Kb + ND;       // (B,H,HD,S)
  __bf16* attnb = Vtb + ND;      // (B,S,D)
  // total: 5*ND + 4*NW = 24M bf16 = 48 MB of workspace

  f32_to_bf16_kernel<<<1024, 256, 0, stream>>>(x, xb, ND);
  f32_to_bf16_kernel<<<512, 256, 0, stream>>>(wq, wqb, NW);
  f32_to_bf16_kernel<<<512, 256, 0, stream>>>(wk, wkb, NW);
  f32_to_bf16_kernel<<<512, 256, 0, stream>>>(wv, wvb, NW);
  f32_to_bf16_kernel<<<512, 256, 0, stream>>>(wo, wob, NW);

  // 128 32-row M-tiles / 4 waves per block = 32 blocks; y = head; z = q/k/v
  qkv_rope_kernel<<<dim3(32, H_, 3), 128, 0, stream>>>(xb, wqb, wkb, wvb, Qb,
                                                       Kb, Vtb);
  // 128 q-tiles / 4 waves per block = 32 blocks; y = head; z = batch
  flash_attn_kernel<<<dim3(S_ / 64, H_, B_), 128, 0, stream>>>(Qb, Kb, Vtb,
                                                               attnb);
  outproj_kernel<<<dim3((B_ * S_) / 128, D_ / 64), 128, 0, stream>>>(
      attnb, wob, (float*)d_out);
}
